// InfoNCELoss_70720931496742
// MI455X (gfx1250) — matrix operand order repaired and checked by reading, not compiled, not run on hardware
//
#include <hip/hip_runtime.h>
#include <hip/hip_bf16.h>
#include <hip/hip_fp16.h>

typedef _Float16 v16h __attribute__((ext_vector_type(16)));
typedef _Float16 v8h  __attribute__((ext_vector_type(8)));
typedef _Float16 v4h  __attribute__((ext_vector_type(4)));
typedef float    v8f  __attribute__((ext_vector_type(8)));
typedef float    v4f  __attribute__((ext_vector_type(4)));

#define DIMK 128
#define INV_T (1.0f / 0.07f)
#define NSEG 8

// ---------------------------------------------------------------------------
// Kernel 1: L2-normalize anchor & positive rows, split each normalized fp32
// value into f16 hi + f16 lo (hi = rn(x), lo = rn(x - hi)), and compute the
// exact fp32 positive similarity pos[i] = dot(a_i, p_i) / T.
// One wave (32 lanes) per row; each lane owns 4 contiguous elements.
// ---------------------------------------------------------------------------
__global__ __launch_bounds__(256) void k_normalize_split(
    const float* __restrict__ anchor, const float* __restrict__ positive,
    _Float16* __restrict__ aH, _Float16* __restrict__ aL,
    _Float16* __restrict__ pH, _Float16* __restrict__ pL,
    float* __restrict__ pos, int B)
{
    const int lane = threadIdx.x & 31;
    const int wave = threadIdx.x >> 5;
    const int row  = blockIdx.x * 8 + wave;
    if (row >= B) return;

    const int base = row * DIMK + lane * 4;
    v4f a = *(const v4f*)(anchor + base);
    v4f p = *(const v4f*)(positive + base);

    float sa = a.x * a.x + a.y * a.y + a.z * a.z + a.w * a.w;
    float sp = p.x * p.x + p.y * p.y + p.z * p.z + p.w * p.w;
#pragma unroll
    for (int m = 1; m < 32; m <<= 1) {
        sa += __shfl_xor(sa, m, 32);
        sp += __shfl_xor(sp, m, 32);
    }
    const float ia = 1.0f / fmaxf(sqrtf(sa), 1e-12f);
    const float ip = 1.0f / fmaxf(sqrtf(sp), 1e-12f);
    a = a * ia;
    p = p * ip;

    float d = a.x * p.x + a.y * p.y + a.z * p.z + a.w * p.w;
#pragma unroll
    for (int m = 1; m < 32; m <<= 1) d += __shfl_xor(d, m, 32);
    if (lane == 0) pos[row] = d * INV_T;

    v4h ah, al, ph, pl;
#pragma unroll
    for (int j = 0; j < 4; ++j) {
        float xa = a[j];
        _Float16 ha = (_Float16)xa;
        ah[j] = ha;
        al[j] = (_Float16)(xa - (float)ha);
        float xp = p[j];
        _Float16 hp = (_Float16)xp;
        ph[j] = hp;
        pl[j] = (_Float16)(xp - (float)hp);
    }
    *(v4h*)(aH + base) = ah;
    *(v4h*)(aL + base) = al;
    *(v4h*)(pH + base) = ph;
    *(v4h*)(pL + base) = pl;
}

// ---------------------------------------------------------------------------
// Kernel 2: fused GEMM (split-f16 WMMA, ~fp32 accuracy) + fixed-shift
// sum-of-exp per row. Each wave owns a 32-row A tile resident in VGPRs
// (2 x 16-row sets x 4 K-chunks x hi/lo = 128 VGPRs) and sweeps 16-wide
// column tiles of its segment. Per tile: 8 b128 operand loads x hi/lo feed
// 24 v_wmma_f32_16x16x32_f16 (48 FLOP/operand-byte), then the 16x(16+16)
// fp32 results get exp((dot-1)/T) accumulated per row. Two independent
// accumulator chains hide WMMA->WMMA RAW latency.
// Grid: (B/256 row-blocks) x (NSEG column segments).
// ---------------------------------------------------------------------------
__global__ __launch_bounds__(256) void k_gemm_rowsum(
    const _Float16* __restrict__ aH, const _Float16* __restrict__ aL,
    const _Float16* __restrict__ pH, const _Float16* __restrict__ pL,
    float* __restrict__ partial, int B, int colsPerSeg)
{
    const int lane     = threadIdx.x & 31;
    const int wave     = threadIdx.x >> 5;
    const int rowBase  = blockIdx.x * 256 + wave * 32;
    const int seg      = blockIdx.y;
    const int colStart = seg * colsPerSeg;

    const int m      = lane & 15;   // matrix row (A) / column (B) within tile
    const int hiHalf = lane >> 4;   // 0: lanes 0-15, 1: lanes 16-31

    // A 16x32 layout (matching B): per-lane 16 contiguous halves at
    // K = kA .. kA+15, kA = hiHalf*16.
    const int kA = hiHalf * 16;
    // B 32x16 layout (code's own k order, shared with A): per-lane 16 contiguous
    // halves at K = kB .. kB+15, kB = hiHalf*16; column = lane&15.
    const int kB = hiHalf * 16;

    union V16 { v16h v; v8h h[2]; };
    V16 Ah[2][4], Al[2][4];
#pragma unroll
    for (int s = 0; s < 2; ++s) {
        const _Float16* arH = aH + (size_t)(rowBase + s * 16 + m) * DIMK;
        const _Float16* arL = aL + (size_t)(rowBase + s * 16 + m) * DIMK;
#pragma unroll
        for (int c = 0; c < 4; ++c) {
            const int k0 = c * 32 + kA;
            Ah[s][c].h[0] = *(const v8h*)(arH + k0);
            Ah[s][c].h[1] = *(const v8h*)(arH + k0 + 8);
            Al[s][c].h[0] = *(const v8h*)(arL + k0);
            Al[s][c].h[1] = *(const v8h*)(arL + k0 + 8);
        }
    }

    float racc0[8] = {0.f, 0.f, 0.f, 0.f, 0.f, 0.f, 0.f, 0.f};
    float racc1[8] = {0.f, 0.f, 0.f, 0.f, 0.f, 0.f, 0.f, 0.f};
    const int tiles = colsPerSeg >> 4;

    for (int t = 0; t < tiles; ++t) {
        const int col = colStart + (t << 4) + m;
        const _Float16* brH = pH + (size_t)col * DIMK;
        const _Float16* brL = pL + (size_t)col * DIMK;

        // prefetch next tile's rows for this lane (speculative, dropped OOB)
        __builtin_prefetch(brH + 16 * DIMK, 0, 1);
        __builtin_prefetch(brL + 16 * DIMK, 0, 1);

        v8f acc0 = {0.f, 0.f, 0.f, 0.f, 0.f, 0.f, 0.f, 0.f};
        v8f acc1 = {0.f, 0.f, 0.f, 0.f, 0.f, 0.f, 0.f, 0.f};
#pragma unroll
        for (int c = 0; c < 4; ++c) {
            const v16h Bh = *(const v16h*)(brH + c * 32 + kB);
            const v16h Bl = *(const v16h*)(brL + c * 32 + kB);
            // hi*hi + hi*lo + lo*hi for both resident row sets; the two
            // acc chains are independent -> no dead hazard slots.
            acc0 = __builtin_amdgcn_wmma_f32_16x16x32_f16(
                false, Ah[0][c].v, false, Bh, (short)0, acc0, false, false);
            acc1 = __builtin_amdgcn_wmma_f32_16x16x32_f16(
                false, Ah[1][c].v, false, Bh, (short)0, acc1, false, false);
            acc0 = __builtin_amdgcn_wmma_f32_16x16x32_f16(
                false, Ah[0][c].v, false, Bl, (short)0, acc0, false, false);
            acc1 = __builtin_amdgcn_wmma_f32_16x16x32_f16(
                false, Ah[1][c].v, false, Bl, (short)0, acc1, false, false);
            acc0 = __builtin_amdgcn_wmma_f32_16x16x32_f16(
                false, Al[0][c].v, false, Bh, (short)0, acc0, false, false);
            acc1 = __builtin_amdgcn_wmma_f32_16x16x32_f16(
                false, Al[1][c].v, false, Bh, (short)0, acc1, false, false);
        }

        // dot products bounded by 1 (normalized rows): fixed shift M0 = 1/T.
#pragma unroll
        for (int r = 0; r < 8; ++r) {
            racc0[r] += __expf((acc0[r] - 1.0f) * INV_T);
            racc1[r] += __expf((acc1[r] - 1.0f) * INV_T);
        }
    }

    // C layout: VGPR r holds row r (lanes 0-15) / row r+8 (lanes 16-31),
    // one column per lane -> reduce across each 16-lane half.
#pragma unroll
    for (int r = 0; r < 8; ++r) {
        float v0 = racc0[r];
        float v1 = racc1[r];
#pragma unroll
        for (int mm = 1; mm < 16; mm <<= 1) {
            v0 += __shfl_xor(v0, mm, 32);
            v1 += __shfl_xor(v1, mm, 32);
        }
        if (m == 0) {
            const int rbase = rowBase + hiHalf * 8 + r;
            partial[(size_t)seg * B + rbase]      = v0;
            partial[(size_t)seg * B + rbase + 16] = v1;
        }
    }
}

// ---------------------------------------------------------------------------
// Kernel 3: per-row loss = M0 + log(sum_exp) - pos[i]; mean over all rows.
// Diagonal term cancels with the positive logit exactly (sim_ii == pos_i),
// so the unmasked row-sum is the full softmax denominator set.
// ---------------------------------------------------------------------------
__global__ __launch_bounds__(256) void k_loss_mean(
    const float* __restrict__ partial, const float* __restrict__ pos,
    float* __restrict__ out, int B)
{
    float local = 0.f;
    for (int i = threadIdx.x; i < B; i += 256) {
        float tot = 0.f;
#pragma unroll
        for (int s = 0; s < NSEG; ++s) tot += partial[(size_t)s * B + i];
        local += INV_T + logf(tot) - pos[i];
    }
    __shared__ float sm[8];
#pragma unroll
    for (int m = 1; m < 32; m <<= 1) local += __shfl_xor(local, m, 32);
    if ((threadIdx.x & 31) == 0) sm[threadIdx.x >> 5] = local;
    __syncthreads();
    if (threadIdx.x == 0) {
        float s = 0.f;
#pragma unroll
        for (int w = 0; w < 8; ++w) s += sm[w];
        out[0] = s / (float)B;
    }
}

extern "C" void kernel_launch(void* const* d_in, const int* in_sizes, int n_in,
                              void* d_out, int out_size, void* d_ws, size_t ws_size,
                              hipStream_t stream) {
    const float* anchor   = (const float*)d_in[0];
    const float* positive = (const float*)d_in[1];
    const int B = in_sizes[0] / DIMK;   // 16384

    // Workspace layout (all offsets 256B-aligned):
    //   aH, aL, pH, pL : B*128 f16 each (4 MB each)
    //   pos            : B f32
    //   partial        : NSEG * B f32
    char* ws = (char*)d_ws;
    const size_t halfBytes = (size_t)B * DIMK * sizeof(_Float16);
    _Float16* aH = (_Float16*)(ws);
    _Float16* aL = (_Float16*)(ws + halfBytes);
    _Float16* pH = (_Float16*)(ws + 2 * halfBytes);
    _Float16* pL = (_Float16*)(ws + 3 * halfBytes);
    float* pos     = (float*)(ws + 4 * halfBytes);
    float* partial = (float*)(ws + 4 * halfBytes + (size_t)B * sizeof(float));

    k_normalize_split<<<B / 8, 256, 0, stream>>>(anchor, positive,
                                                 aH, aL, pH, pL, pos, B);

    dim3 grid2(B / 256, NSEG);
    k_gemm_rowsum<<<grid2, 256, 0, stream>>>(aH, aL, pH, pL, partial, B, B / NSEG);

    k_loss_mean<<<1, 256, 0, stream>>>(partial, pos, (float*)d_out, B);
}
